// GraphConvModel_WithEdgeFeat_30743375905246
// MI455X (gfx1250) — matrix operand order repaired and checked by reading, not compiled, not run on hardware
//
#include <hip/hip_runtime.h>

typedef __attribute__((ext_vector_type(16))) _Float16 v16h;
typedef __attribute__((ext_vector_type(8)))  float    v8f;

#define N_FEAT   64
#define N_CLS    10

// ---------------------------------------------------------------------------
// Weight convert: w (64x64 f32, [K][N]) -> wt (64x64 f16, transposed [N][K])
// ---------------------------------------------------------------------------
__global__ __launch_bounds__(256) void convert_w_kernel(const float* __restrict__ w,
                                                        _Float16* __restrict__ wt) {
    int t = blockIdx.x * 256 + threadIdx.x;      // 0..4095
    if (t >= N_FEAT * N_FEAT) return;
    int n = t >> 6;          // output row (N)
    int k = t & 63;          // output col (K)
    wt[t] = (_Float16)w[k * N_FEAT + n];
}

// ---------------------------------------------------------------------------
// Edge scatter: agg[dst[e]][:] += ew[e] * x[src[e]][:]
// One wave per edge, each lane handles 2 consecutive features.
// ---------------------------------------------------------------------------
__global__ __launch_bounds__(256) void scatter_edges_kernel(
    const float* __restrict__ x, const float* __restrict__ ew,
    const int* __restrict__ src, const int* __restrict__ dst,
    float* __restrict__ agg, int nEdges) {
    unsigned long long t = (unsigned long long)blockIdx.x * 256ull + threadIdx.x;
    int e = (int)(t >> 5);
    int f = ((int)t & 31) * 2;
    if (e >= nEdges) return;
    int s = src[e];
    int d = dst[e];
    float w = ew[e];
    const float2 v = *(const float2*)(x + (size_t)s * N_FEAT + f);
    atomicAdd(agg + (size_t)d * N_FEAT + f,     v.x * w);
    atomicAdd(agg + (size_t)d * N_FEAT + f + 1, v.y * w);
}

// ---------------------------------------------------------------------------
// Fused dual GEMM: out = [relu]( agg @ Wrel + xin @ Wroot + b )
// A: f32 in memory, converted to f16 in registers. Weights pre-transposed f16.
// One wave -> 16 node rows x all 64 output cols (4 col tiles, 16 WMMAs).
// 8 waves / block -> 128 rows per block.
// ---------------------------------------------------------------------------
__global__ __launch_bounds__(256) void gemm_layer_kernel(
    const float* __restrict__ agg, const float* __restrict__ xin,
    const _Float16* __restrict__ WtRel, const _Float16* __restrict__ WtRoot,
    const float* __restrict__ bias, float* __restrict__ out,
    int nNodes, int doRelu) {
    const int wave  = threadIdx.x >> 5;
    const int lane  = threadIdx.x & 31;
    const int m     = lane & 15;        // A-fragment row within tile / B,D column
    const int khalf = lane >> 4;        // lane group 0/1

    const int rowBase = (blockIdx.x * 8 + wave) * 16;
    int row = rowBase + m;
    int rowC = row < nNodes ? row : (nNodes - 1);   // clamp: keep EXEC all-1s

    // ---- Load A fragments (2 K-chunks x 2 matrices), cvt f32->f16 ----
    // element e in 0..7  -> K = kc*32 + khalf*16 + e
    // element e in 8..15 -> K = kc*32 + khalf*16 + 8 + (e-8)
    v16h aAgg[2], aX[2];
    const float* pa = agg + (size_t)rowC * N_FEAT;
    const float* px = xin + (size_t)rowC * N_FEAT;
#pragma unroll
    for (int kc = 0; kc < 2; ++kc) {
        int k0 = kc * 32 + khalf * 16;
        int k1 = k0 + 8;
#pragma unroll
        for (int e = 0; e < 8; ++e) {
            aAgg[kc][e]     = (_Float16)pa[k0 + e];
            aAgg[kc][8 + e] = (_Float16)pa[k1 + e];
            aX[kc][e]       = (_Float16)px[k0 + e];
            aX[kc][8 + e]   = (_Float16)px[k1 + e];
        }
    }

    // ---- 4 column tiles of 16 ----
#pragma unroll
    for (int nt = 0; nt < 4; ++nt) {
        const int col = nt * 16 + m;
        v8f acc = {};
#pragma unroll
        for (int kc = 0; kc < 2; ++kc) {
            // B fragment: element e -> K = kc*32 + khalf*16 + e (contiguous 16)
            const v16h bRel  = *(const v16h*)(WtRel  + (size_t)col * N_FEAT + kc * 32 + khalf * 16);
            acc = __builtin_amdgcn_wmma_f32_16x16x32_f16(
                false, aAgg[kc], false, bRel, (short)0, acc, false, false);
            const v16h bRoot = *(const v16h*)(WtRoot + (size_t)col * N_FEAT + kc * 32 + khalf * 16);
            acc = __builtin_amdgcn_wmma_f32_16x16x32_f16(
                false, aX[kc], false, bRoot, (short)0, acc, false, false);
        }
        const float bv = bias[col];
        // D layout: VGPR v -> row = rowBase + khalf*8 + v, col per lane
#pragma unroll
        for (int v = 0; v < 8; ++v) {
            int r = rowBase + khalf * 8 + v;
            if (r < nNodes) {
                float val = acc[v] + bv;
                if (doRelu) val = fmaxf(val, 0.0f);
                out[(size_t)r * N_FEAT + col] = val;
            }
        }
    }
}

// ---------------------------------------------------------------------------
// Pool scatter: sums[batch[n]][:] += h[n][:]; counts[batch[n]] += 1
// ---------------------------------------------------------------------------
__global__ __launch_bounds__(256) void pool_scatter_kernel(
    const float* __restrict__ h, const int* __restrict__ batch,
    float* __restrict__ sums, float* __restrict__ counts, int nNodes) {
    unsigned long long t = (unsigned long long)blockIdx.x * 256ull + threadIdx.x;
    int node = (int)(t >> 5);
    int f = ((int)t & 31) * 2;
    if (node >= nNodes) return;
    int g = batch[node];
    const float2 v = *(const float2*)(h + (size_t)node * N_FEAT + f);
    atomicAdd(sums + (size_t)g * N_FEAT + f,     v.x);
    atomicAdd(sums + (size_t)g * N_FEAT + f + 1, v.y);
    if (f == 0) atomicAdd(counts + g, 1.0f);
}

// ---------------------------------------------------------------------------
// Head: pooled = sums/max(counts,1); logits = pooled @ lin_w + lin_b; softmax
// One thread per graph.
// ---------------------------------------------------------------------------
__global__ __launch_bounds__(128) void head_kernel(
    const float* __restrict__ sums, const float* __restrict__ counts,
    const float* __restrict__ lin_w, const float* __restrict__ lin_b,
    float* __restrict__ out, int nGraphs) {
    int g = blockIdx.x * 128 + threadIdx.x;
    if (g >= nGraphs) return;
    float inv = 1.0f / fmaxf(counts[g], 1.0f);
    float logits[N_CLS];
#pragma unroll
    for (int j = 0; j < N_CLS; ++j) logits[j] = lin_b[j];
    for (int k = 0; k < N_FEAT; ++k) {
        float p = sums[(size_t)g * N_FEAT + k] * inv;
#pragma unroll
        for (int j = 0; j < N_CLS; ++j) logits[j] += p * lin_w[k * N_CLS + j];
    }
    float mx = logits[0];
#pragma unroll
    for (int j = 1; j < N_CLS; ++j) mx = fmaxf(mx, logits[j]);
    float s = 0.0f;
    float ex[N_CLS];
#pragma unroll
    for (int j = 0; j < N_CLS; ++j) { ex[j] = __expf(logits[j] - mx); s += ex[j]; }
    float rs = 1.0f / s;
#pragma unroll
    for (int j = 0; j < N_CLS; ++j) out[(size_t)g * N_CLS + j] = ex[j] * rs;
}

// ---------------------------------------------------------------------------
// Launch
// ---------------------------------------------------------------------------
extern "C" void kernel_launch(void* const* d_in, const int* in_sizes, int n_in,
                              void* d_out, int out_size, void* d_ws, size_t ws_size,
                              hipStream_t stream) {
    const float* x    = (const float*)d_in[0];
    const float* ew   = (const float*)d_in[1];
    const int*   ei   = (const int*)d_in[2];
    const int*   bat  = (const int*)d_in[3];
    const float* wrel[3] = { (const float*)d_in[4], (const float*)d_in[7], (const float*)d_in[10] };
    const float* wrot[3] = { (const float*)d_in[5], (const float*)d_in[8], (const float*)d_in[11] };
    const float* bia[3]  = { (const float*)d_in[6], (const float*)d_in[9], (const float*)d_in[12] };
    const float* lin_w = (const float*)d_in[13];
    const float* lin_b = (const float*)d_in[14];
    float* out = (float*)d_out;

    const int nNodes  = in_sizes[3];
    const int nEdges  = in_sizes[1];
    const int nGraphs = out_size / N_CLS;
    const int* src = ei;
    const int* dst = ei + nEdges;

    // Workspace carve-up (256B aligned). ~38.5 MB total.
    char* ws = (char*)d_ws;
    size_t off = 0;
    auto carve = [&](size_t bytes) -> char* {
        char* p = ws + off;
        off = (off + bytes + 255) & ~(size_t)255;
        return p;
    };
    const size_t featBytes = (size_t)nNodes * N_FEAT * sizeof(float);
    float*    agg   = (float*)carve(featBytes);
    float*    hA    = (float*)carve(featBytes);
    float*    hB    = (float*)carve(featBytes);
    _Float16* wt    = (_Float16*)carve(6 * N_FEAT * N_FEAT * sizeof(_Float16));
    float*    sums  = (float*)carve((size_t)nGraphs * N_FEAT * sizeof(float));
    float*    cnts  = (float*)carve((size_t)nGraphs * sizeof(float));
    (void)ws_size; (void)n_in;

    _Float16* wtRel[3] = { wt + 0 * 4096, wt + 2 * 4096, wt + 4 * 4096 };
    _Float16* wtRot[3] = { wt + 1 * 4096, wt + 3 * 4096, wt + 5 * 4096 };

    // Convert all 6 weight matrices to transposed f16 (tiny, runs each call).
    for (int l = 0; l < 3; ++l) {
        convert_w_kernel<<<16, 256, 0, stream>>>(wrel[l], wtRel[l]);
        convert_w_kernel<<<16, 256, 0, stream>>>(wrot[l], wtRot[l]);
    }

    const unsigned scatterBlocks = (unsigned)(((unsigned long long)nEdges * 32ull + 255ull) / 256ull);
    const unsigned gemmBlocks    = (unsigned)((nNodes + 127) / 128);
    const unsigned poolBlocks    = (unsigned)(((unsigned long long)nNodes * 32ull + 255ull) / 256ull);

    // ---- Layer 1: x -> hA ----
    hipMemsetAsync(agg, 0, featBytes, stream);
    scatter_edges_kernel<<<scatterBlocks, 256, 0, stream>>>(x, ew, src, dst, agg, nEdges);
    gemm_layer_kernel<<<gemmBlocks, 256, 0, stream>>>(agg, x, wtRel[0], wtRot[0], bia[0], hA, nNodes, 1);

    // ---- Layer 2: hA -> hB ----
    hipMemsetAsync(agg, 0, featBytes, stream);
    scatter_edges_kernel<<<scatterBlocks, 256, 0, stream>>>(hA, ew, src, dst, agg, nEdges);
    gemm_layer_kernel<<<gemmBlocks, 256, 0, stream>>>(agg, hA, wtRel[1], wtRot[1], bia[1], hB, nNodes, 1);

    // ---- Layer 3: hB -> hA (no relu) ----
    hipMemsetAsync(agg, 0, featBytes, stream);
    scatter_edges_kernel<<<scatterBlocks, 256, 0, stream>>>(hB, ew, src, dst, agg, nEdges);
    gemm_layer_kernel<<<gemmBlocks, 256, 0, stream>>>(agg, hB, wtRel[2], wtRot[2], bia[2], hA, nNodes, 0);

    // ---- Global mean pool + head ----
    hipMemsetAsync(sums, 0, (size_t)nGraphs * N_FEAT * sizeof(float), stream);
    hipMemsetAsync(cnts, 0, (size_t)nGraphs * sizeof(float), stream);
    pool_scatter_kernel<<<poolBlocks, 256, 0, stream>>>(hA, bat, sums, cnts, nNodes);
    head_kernel<<<(nGraphs + 127) / 128, 128, 0, stream>>>(sums, cnts, lin_w, lin_b, out, nGraphs);
}
